// MultiHeadedAttention_43585328120242
// MI455X (gfx1250) — matrix operand order repaired and checked by reading, not compiled, not run on hardware
//
#include <hip/hip_runtime.h>
#include <hip/hip_bf16.h>

// ---------------------------------------------------------------------------
// MultiHeadedAttention for MI455X (gfx1250), wave32 + WMMA f32_16x16x32_f16.
//
// Pipeline:
//   1) proj_kernel   x3 : Q/K/V = W*X + b  (f32 in, f16 out; Q pre-scaled 1/8)
//        Q,K layout: [b,h,n,64]   V layout: [b,h,64,N]
//   2) attn_kernel       : flash attention, 4 waves/block (64 query rows),
//        K/V tiles staged once per block into LDS, P staged per-wave in LDS.
//        out layout: [b,n,256] f16
//   3) oproj_kernel      : out = Wm*X + bm  (f16 in, f32 out) [b,256,2048]
// ---------------------------------------------------------------------------

#define D_MODEL  256
#define NHEADS   4
#define HDIM     64
#define BATCH    4
#define SEQ      2048

typedef __attribute__((ext_vector_type(16))) _Float16 v16h;
typedef __attribute__((ext_vector_type(8)))  _Float16 v8h;
typedef __attribute__((ext_vector_type(8)))  float    v8f;

__device__ __forceinline__ v8f wmma16(v16h a, v16h b, v8f c) {
  return __builtin_amdgcn_wmma_f32_16x16x32_f16(
      /*neg_a=*/false, a, /*neg_b=*/false, b,
      /*c_mod=*/(short)0, c, /*reuse_a=*/false, /*reuse_b=*/false);
}

// A-operand fragment (16 M x 32 K), source is row-major [row][ldk] f16.
// Lane L: row = L&15, group g = L>>4.
//   elements 0..7  -> K = k0 + g*16 + j       (16B contiguous)
//   elements 8..15 -> K = k0 + g*16 + 8 + j   (16B contiguous)
__device__ __forceinline__ v16h load_a_frag(const _Float16* p, int ldk, int k0) {
  const int lane = threadIdx.x & 31;
  const int r = lane & 15, g = lane >> 4;
  const _Float16* q0 = p + (size_t)r * ldk + k0 + g * 16;
  v8h lo = *(const v8h*)q0;
  v8h hi = *(const v8h*)(q0 + 8);
  v16h out;
  #pragma unroll
  for (int i = 0; i < 8; ++i) { out[i] = lo[i]; out[i + 8] = hi[i]; }
  return out;
}

// A-operand fragment from f32 row-major [row][ldk] (converted to f16).
__device__ __forceinline__ v16h load_a_frag_f32(const float* p, int ldk, int k0) {
  const int lane = threadIdx.x & 31;
  const int r = lane & 15, g = lane >> 4;
  const float* q0 = p + (size_t)r * ldk + k0 + g * 16;
  v16h out;
  #pragma unroll
  for (int i = 0; i < 8; ++i) {
    out[i]     = (_Float16)q0[i];
    out[i + 8] = (_Float16)q0[8 + i];
  }
  return out;
}

// B-operand fragment (32 K x 16 N), source stored as B^T row-major:
// mem[col][k] with row length ldk. Lane L: col = L&15, elems K = k0 + (L>>4)*16 + e
// -> one contiguous 32B load per lane.
__device__ __forceinline__ v16h load_bt_frag(const _Float16* p, int ldk, int k0) {
  const int lane = threadIdx.x & 31;
  const int c = lane & 15, g = lane >> 4;
  return *(const v16h*)(p + (size_t)c * ldk + k0 + g * 16);
}

// ---------------------------------------------------------------------------
// Kernel 1: pointwise-conv projection.  out = scale * (W @ X + bias), f16 out.
// grid = (SEQ/64, D_MODEL/64, BATCH), block = 128 (4 waves).
// layout 0: out[((b*4 + o/64)*SEQ + n)*64 + (o%64)]   (Q, K : [b,h,n,d])
// layout 1: out[((b*4 + o/64)*64 + (o%64))*SEQ + n]   (V    : [b,h,d,n])
// ---------------------------------------------------------------------------
__global__ __launch_bounds__(128) void proj_kernel(
    const float* __restrict__ X,     // [BATCH][D_MODEL][SEQ]
    const float* __restrict__ W,     // [D_MODEL][D_MODEL]
    const float* __restrict__ bias,  // [D_MODEL]
    _Float16* __restrict__ out,
    int layout, float scale)
{
  const int nblk = blockIdx.x * 64;
  const int oblk = blockIdx.y * 64;
  const int b    = blockIdx.z;
  const int wid  = threadIdx.x >> 5;
  const int lane = threadIdx.x & 31;
  const int otile = oblk + wid * 16;
  const int g = lane >> 4;

  __shared__ _Float16 xs[64][48];   // transposed tile [n][c], 96B row stride

  const float* Xb = X + (size_t)b * D_MODEL * SEQ;

  v8f acc[4];
  #pragma unroll
  for (int t = 0; t < 4; ++t)
    #pragma unroll
    for (int j = 0; j < 8; ++j)
      acc[t][j] = bias[otile + j + 8 * g];

  for (int kk = 0; kk < D_MODEL; kk += 32) {
    __syncthreads();
    #pragma unroll
    for (int i = 0; i < 16; ++i) {
      int idx = threadIdx.x + i * 128;       // 0..2047 over 32c x 64n tile
      int cc = idx >> 6;
      int nn = idx & 63;
      xs[nn][cc] = (_Float16)Xb[(size_t)(kk + cc) * SEQ + nblk + nn];
    }
    __syncthreads();

    v16h af = load_a_frag_f32(W + (size_t)otile * D_MODEL, D_MODEL, kk);
    #pragma unroll
    for (int t = 0; t < 4; ++t) {
      v16h bf = *(const v16h*)&xs[t * 16 + (lane & 15)][g * 16];
      acc[t] = wmma16(af, bf, acc[t]);
    }
  }

  #pragma unroll
  for (int t = 0; t < 4; ++t) {
    #pragma unroll
    for (int j = 0; j < 8; ++j) {
      int o = otile + j + 8 * g;
      int n = nblk + t * 16 + (lane & 15);
      float v = acc[t][j] * scale;
      size_t idx;
      if (layout == 0)
        idx = ((size_t)(b * NHEADS + (o >> 6)) * SEQ + n) * HDIM + (o & 63);
      else
        idx = ((size_t)(b * NHEADS + (o >> 6)) * HDIM + (o & 63)) * SEQ + n;
      out[idx] = (_Float16)v;
    }
  }
}

// ---------------------------------------------------------------------------
// Kernel 2: flash attention.
// grid = (SEQ/64, NHEADS, BATCH), block = 128 (4 waves, 16 query rows each).
// K/V tiles for each 32-column m-step are staged once per block into LDS and
// shared by all 4 waves; next tile is prefetched into L2 during compute.
// Q pre-scaled by 1/sqrt(HDIM).  Writes O as [b][n][256] f16.
// ---------------------------------------------------------------------------
__global__ __launch_bounds__(128) void attn_kernel(
    const _Float16* __restrict__ Q,   // [b,h,n,64]
    const _Float16* __restrict__ K,   // [b,h,m,64]
    const _Float16* __restrict__ V,   // [b,h,64,SEQ]
    _Float16* __restrict__ O)         // [b][SEQ][256]
{
  const int h = blockIdx.y;
  const int b = blockIdx.z;
  const int wid  = threadIdx.x >> 5;
  const int lane = threadIdx.x & 31;
  const int g = lane >> 4;
  const int nbase = blockIdx.x * 64 + wid * 16;
  const int bh = b * NHEADS + h;

  const _Float16* Qbh = Q + (size_t)bh * SEQ * HDIM;
  const _Float16* Kbh = K + (size_t)bh * SEQ * HDIM;
  const _Float16* Vbh = V + (size_t)bh * HDIM * SEQ;

  __shared__ _Float16 Ks[32][80];        // K tile [m_local][d], 160B stride
  __shared__ _Float16 Vs[64][48];        // V tile [d][m_local], 96B stride
  __shared__ _Float16 ps[4][16][48];     // per-wave P tile [n_local][m_local]

  // Q A-fragments for this wave's 16-row tile (registers for whole sweep).
  const v16h qa0 = load_a_frag(Qbh + (size_t)nbase * HDIM, HDIM, 0);
  const v16h qa1 = load_a_frag(Qbh + (size_t)nbase * HDIM, HDIM, 32);

  v8f o_acc[4];
  #pragma unroll
  for (int t = 0; t < 4; ++t)
    #pragma unroll
    for (int j = 0; j < 8; ++j) o_acc[t][j] = 0.0f;

  float mrow[8], lrow[8];
  #pragma unroll
  for (int j = 0; j < 8; ++j) { mrow[j] = -1.0e30f; lrow[j] = 0.0f; }

  for (int m0 = 0; m0 < SEQ; m0 += 32) {
    // ---- cooperative staging of K (32x64) and V (64x32) tiles into LDS ----
    __syncthreads();   // WAR: everyone done reading previous Ks/Vs
    #pragma unroll
    for (int i = 0; i < 2; ++i) {
      int idx = threadIdx.x + i * 128;     // 0..255 : 16B chunks of K tile
      int row = idx >> 3, seg = idx & 7;
      *(v8h*)&Ks[row][seg * 8] =
          *(const v8h*)(Kbh + (size_t)(m0 + row) * HDIM + seg * 8);
    }
    #pragma unroll
    for (int i = 0; i < 2; ++i) {
      int idx = threadIdx.x + i * 128;     // 0..255 : 16B chunks of V tile
      int row = idx >> 2, seg = idx & 3;
      *(v8h*)&Vs[row][seg * 8] =
          *(const v8h*)(Vbh + (size_t)row * SEQ + m0 + seg * 8);
    }
    __syncthreads();

    // L2 prefetch of next tile while we compute on this one.
    if (m0 + 32 < SEQ) {
      __builtin_prefetch(Kbh + (size_t)(m0 + 32 + (threadIdx.x >> 2)) * HDIM, 0, 3);
      __builtin_prefetch(Vbh + (size_t)(threadIdx.x >> 1) * SEQ + m0 + 32, 0, 3);
    }

    // ---- S = Q^T K for two 16-col m-subtiles (K-dim = HDIM = 2 x 32) ----
    v8f s0 = {}, s1 = {};
    s0 = wmma16(qa0, load_bt_frag(&Ks[0][0],  80, 0),  s0);
    s0 = wmma16(qa1, load_bt_frag(&Ks[0][0],  80, 32), s0);
    s1 = wmma16(qa0, load_bt_frag(&Ks[16][0], 80, 0),  s1);
    s1 = wmma16(qa1, load_bt_frag(&Ks[16][0], 80, 32), s1);

    // ---- online softmax; rows live in half-wave lane groups ----
    #pragma unroll
    for (int j = 0; j < 8; ++j) {
      float mx = fmaxf(s0[j], s1[j]);
      #pragma unroll
      for (int off = 8; off >= 1; off >>= 1)
        mx = fmaxf(mx, __shfl_xor(mx, off, 32));
      float mnew = fmaxf(mrow[j], mx);
      float corr = __expf(mrow[j] - mnew);
      float p0 = __expf(s0[j] - mnew);
      float p1 = __expf(s1[j] - mnew);
      float rs = p0 + p1;
      #pragma unroll
      for (int off = 8; off >= 1; off >>= 1)
        rs += __shfl_xor(rs, off, 32);
      lrow[j] = lrow[j] * corr + rs;
      mrow[j] = mnew;
      #pragma unroll
      for (int t = 0; t < 4; ++t) o_acc[t][j] *= corr;

      const int r = j + 8 * g;               // fragment row -> n_local
      ps[wid][r][(lane & 15)]      = (_Float16)p0;
      ps[wid][r][16 + (lane & 15)] = (_Float16)p1;
    }
    // P write -> read is same-wave LDS traffic: architecturally in order.

    // ---- O += P * V^T  (A = P [16n x 32m], B = V^T [32m x 16d]) ----
    v16h pa = load_a_frag(&ps[wid][0][0], 48, 0);
    #pragma unroll
    for (int t = 0; t < 4; ++t) {
      v16h vb = load_bt_frag(&Vs[t * 16][0], 48, 0);
      o_acc[t] = wmma16(pa, vb, o_acc[t]);
    }
  }

  // ---- normalize and store [b][n][c] ----
  _Float16* Ob = O + (size_t)b * SEQ * D_MODEL;
  #pragma unroll
  for (int t = 0; t < 4; ++t) {
    #pragma unroll
    for (int j = 0; j < 8; ++j) {
      int n = nbase + j + 8 * g;
      int c = h * HDIM + t * 16 + (lane & 15);
      Ob[(size_t)n * D_MODEL + c] = (_Float16)(o_acc[t][j] / lrow[j]);
    }
  }
}

// ---------------------------------------------------------------------------
// Kernel 3: output projection.  out = Wm @ X + bm, f32 out [b][256][2048].
// X is [b][n][256] f16 so the B operand is contiguous along K=c.
// grid = (SEQ/64, D_MODEL/64, BATCH), block = 128.
// ---------------------------------------------------------------------------
__global__ __launch_bounds__(128) void oproj_kernel(
    const _Float16* __restrict__ Xt,  // [b][SEQ][256]
    const float* __restrict__ W,      // [256][256]
    const float* __restrict__ bias,   // [256]
    float* __restrict__ out)          // [b][256][SEQ]
{
  const int nblk = blockIdx.x * 64;
  const int oblk = blockIdx.y * 64;
  const int b    = blockIdx.z;
  const int wid  = threadIdx.x >> 5;
  const int lane = threadIdx.x & 31;
  const int otile = oblk + wid * 16;
  const int g = lane >> 4;

  const _Float16* Xb = Xt + (size_t)b * SEQ * D_MODEL;

  v8f acc[4];
  #pragma unroll
  for (int t = 0; t < 4; ++t)
    #pragma unroll
    for (int j = 0; j < 8; ++j)
      acc[t][j] = bias[otile + j + 8 * g];

  for (int kk = 0; kk < D_MODEL; kk += 32) {
    v16h af = load_a_frag_f32(W + (size_t)otile * D_MODEL, D_MODEL, kk);
    #pragma unroll
    for (int t = 0; t < 4; ++t) {
      v16h bf = load_bt_frag(Xb + (size_t)(nblk + t * 16) * D_MODEL, D_MODEL, kk);
      acc[t] = wmma16(af, bf, acc[t]);
    }
  }

  float* Outb = out + (size_t)b * D_MODEL * SEQ;
  #pragma unroll
  for (int t = 0; t < 4; ++t) {
    #pragma unroll
    for (int j = 0; j < 8; ++j) {
      int o = otile + j + 8 * g;
      int n = nblk + t * 16 + (lane & 15);
      Outb[(size_t)o * SEQ + n] = acc[t][j];
    }
  }
}

// ---------------------------------------------------------------------------
extern "C" void kernel_launch(void* const* d_in, const int* in_sizes, int n_in,
                              void* d_out, int out_size, void* d_ws, size_t ws_size,
                              hipStream_t stream) {
  (void)in_sizes; (void)n_in; (void)out_size; (void)ws_size;

  const float* query = (const float*)d_in[0];
  const float* key   = (const float*)d_in[1];
  const float* value = (const float*)d_in[2];
  const float* wq = (const float*)d_in[3];
  const float* bq = (const float*)d_in[4];
  const float* wk = (const float*)d_in[5];
  const float* bk = (const float*)d_in[6];
  const float* wv = (const float*)d_in[7];
  const float* bv = (const float*)d_in[8];
  const float* wm = (const float*)d_in[9];
  const float* bm = (const float*)d_in[10];

  const size_t ELEMS = (size_t)BATCH * D_MODEL * SEQ;  // 2,097,152 per tensor
  _Float16* qf = (_Float16*)d_ws;
  _Float16* kf = qf + ELEMS;
  _Float16* vf = kf + ELEMS;
  _Float16* xf = vf + ELEMS;   // attention output [b][n][256]

  dim3 gridP(SEQ / 64, D_MODEL / 64, BATCH);
  const float qscale = 0.125f;  // 1/sqrt(HDIM)

  proj_kernel<<<gridP, 128, 0, stream>>>(query, wq, bq, qf, 0, qscale);
  proj_kernel<<<gridP, 128, 0, stream>>>(key,   wk, bk, kf, 0, 1.0f);
  proj_kernel<<<gridP, 128, 0, stream>>>(value, wv, bv, vf, 1, 1.0f);

  dim3 gridA(SEQ / 64, NHEADS, BATCH);
  attn_kernel<<<gridA, 128, 0, stream>>>(qf, kf, vf, xf);

  oproj_kernel<<<gridP, 128, 0, stream>>>(xf, wm, bm, (float*)d_out);
}
